// SpatialGRU_86363202388272
// MI455X (gfx1250) — matrix operand order repaired and checked by reading, not compiled, not run on hardware
//
#include <hip/hip_runtime.h>
#include <hip/hip_bf16.h>

// ---------------- problem constants ----------------
static constexpr int BB = 256;   // batch
static constexpr int CC = 32;    // channels
static constexpr int LL = 64;    // rows
static constexpr int RR = 64;    // cols
static constexpr int UU = 128;   // units
static constexpr int D1 = 416;   // K of GEMM1 = C + 3*U
static constexpr int NN = 1024;  // N of fused GEMM1 = 384(r) + 512(z) + 128(w_ij)
static constexpr int K2 = 384;   // K of GEMM2 = 3*U

static constexpr int CELL  = BB * UU;        // 32768 floats, one h[col] slab
static constexpr int ROWSZ = RR * CELL;      // one h row buffer (floats)

// ---------------- workspace layout (bytes) ----------------
static constexpr size_t OFF_WT = 0;                                   // f16 [1024][416]
static constexpr size_t OFF_UT = 864ull * 1024;                       // f16 [128][384]
static constexpr size_t OFF_HB = 1ull << 20;                          // f32 3 row buffers
static constexpr size_t OFF_RC = OFF_HB + 3ull * ROWSZ * 4;           // f16 [64][256][384]
static constexpr size_t OFF_Z  = OFF_RC + 64ull * BB * K2 * 2;        // f32 [64][4][256][128]
static constexpr size_t OFF_MI = OFF_Z  + 64ull * 4 * CELL * 4;       // f32 [64][256][128]
static constexpr size_t OFF_QF = OFF_MI + 64ull * CELL * 4;           // f16 [64][256][416]

typedef _Float16 h16 __attribute__((ext_vector_type(16)));
typedef _Float16 h8  __attribute__((ext_vector_type(8)));
typedef _Float16 h4  __attribute__((ext_vector_type(4)));
typedef float    f8  __attribute__((ext_vector_type(8)));

#define DEV __device__ __forceinline__

DEV f8 wmma32(h16 a, h16 b, f8 c) {
  // D = A(16x32 f16) * B(32x16 f16) + C(16x16 f32)
  return __builtin_amdgcn_wmma_f32_16x16x32_f16(false, a, false, b, (short)0, c,
                                                false, false);
}

// A-operand fragment (16x32 f16 layout): lane L -> row L%16,
// halves 0..15 = K[(L/16)*16 .. +15] contiguous
DEV h16 ldfragA(const _Float16* p) {
  h16 f;
  *(h8*)&f       = *(const h8*)p;
  *((h8*)&f + 1) = *(const h8*)(p + 8);
  return f;
}
// B-operand fragment (32x16 f16): lane L -> col L%16,
// halves 0..15 = K[(L/16)*16 .. +15] contiguous (32B aligned).
DEV h16 ldfragB(const _Float16* p) { return *(const h16*)p; }

DEV float sigm(float x) { return 1.0f / (1.0f + __expf(-x)); }

// ---------------- weight prep: fused transposed f16 weights ----------------
__global__ void k_prep_wt(const float* __restrict__ wr, const float* __restrict__ wz,
                          const float* __restrict__ wij, _Float16* __restrict__ WT) {
  int idx = blockIdx.x * 256 + threadIdx.x;
  if (idx >= NN * D1) return;
  int n = idx / D1, k = idx - n * D1;
  float v;
  if (n < 384)       v = wr[k * 384 + n];
  else if (n < 896)  v = wz[k * 512 + (n - 384)];
  else               v = (k >= 384) ? wij[(k - 384) * UU + (n - 896)] : 0.0f;
  WT[idx] = (_Float16)v;
}

__global__ void k_prep_ut(const float* __restrict__ U, _Float16* __restrict__ UT) {
  int idx = blockIdx.x * 256 + threadIdx.x;
  if (idx >= UU * K2) return;
  int n = idx / K2, k = idx - n * K2;
  UT[idx] = (_Float16)U[k * UU + n];
}

// ---------------- kernel Q: stage q = [h_top|h_left|h_diag|s_ij] as f16 ----------
// grid: x = quarter (64 rows), y = cell. One pass per diagonal; x-gather done once.
__global__ __launch_bounds__(256) void k_stageQ(
    const float* __restrict__ x, const float* __restrict__ hbuf,
    _Float16* __restrict__ QF, int d, int i0) {
  const int cell = blockIdx.y;
  const int i = i0 + cell, j = d - i;
  const int m0 = blockIdx.x * 64;
  const int tid = threadIdx.x;
  const bool hasT = (i > 0), hasL = (j > 0), hasD = hasT && hasL;
  const int jm1 = hasL ? (j - 1) : 0;
  const float* pT = hbuf + (size_t)((i + 2) % 3) * ROWSZ + (size_t)j   * CELL;
  const float* pL = hbuf + (size_t)( i      % 3) * ROWSZ + (size_t)jm1 * CELL;
  const float* pD = hbuf + (size_t)((i + 2) % 3) * ROWSZ + (size_t)jm1 * CELL;
  _Float16* Qc = QF + (size_t)cell * BB * D1;

  // h parts: 64 rows x 384 cols as float4 -> h4 (6144 groups, 24/thread)
  for (int it = 0; it < 24; ++it) {
    int idx = it * 256 + tid;
    int m = m0 + idx / 96;
    int k4 = (idx % 96) * 4;
    int g = k4 >> 7, kk = k4 & 127;
    const float* src = (g == 0) ? pT : (g == 1) ? pL : pD;
    const bool ok = (g == 0) ? hasT : (g == 1) ? hasL : hasD;
    float4 v = ok ? *(const float4*)(src + m * UU + kk)
                  : make_float4(0.f, 0.f, 0.f, 0.f);
    h4 hv = {(_Float16)v.x, (_Float16)v.y, (_Float16)v.z, (_Float16)v.w};
    *(h4*)&Qc[(size_t)m * D1 + k4] = hv;
  }
  // s_ij part: 64 rows x 32 channels (strided gather, once per cell)
  for (int it = 0; it < 8; ++it) {
    int idx = it * 256 + tid;
    int m = m0 + (idx >> 5), c = idx & 31;
    Qc[(size_t)m * D1 + 384 + c] =
        (_Float16)x[(((size_t)m * CC + c) * LL + i) * RR + j];
  }
}

// ---------------- kernel A: GEMM1 + gates per cell (pure WMMA, no LDS) --------
// grid: x = strip s in [0,8) (16-unit column group), y = cell on diagonal.
// Each wave: 2 M-tiles x 8 N-tiles; N-tile t covers columns t*128 + s*16 + [0,16).
// t=0..2 -> r groups, t=3..6 -> z groups (zi,zl,zt,zd), t=7 -> s@w_ij.
__global__ __launch_bounds__(256) void k_cellA(
    const _Float16* __restrict__ QF, const _Float16* __restrict__ WT,
    const float* __restrict__ br, const float* __restrict__ bz,
    const float* __restrict__ hbuf,
    _Float16* __restrict__ RC, float* __restrict__ Z, float* __restrict__ MI,
    int d, int i0) {
  const int cell = blockIdx.y;
  const int i = i0 + cell, j = d - i;
  const int s = blockIdx.x;
  const int tid = threadIdx.x, lane = tid & 31, wave = tid >> 5;
  const int mt0 = wave * 2;
  const int hs = lane >> 4, ucl = lane & 15;

  const _Float16* QFc = QF + (size_t)cell * BB * D1;
  const _Float16* aBase0 = QFc + (size_t)((mt0    ) * 16 + ucl) * D1 + hs * 16;
  const _Float16* aBase1 = QFc + (size_t)((mt0 + 1) * 16 + ucl) * D1 + hs * 16;
  const _Float16* bBase  = WT + (size_t)(s * 16 + ucl) * D1 + hs * 16;

  f8 acc[2][8] = {};
#pragma unroll 1
  for (int kc = 0; kc < 13; ++kc) {
    h16 a0 = ldfragA(aBase0 + kc * 32);
    h16 a1 = ldfragA(aBase1 + kc * 32);
    h16 b[8];
#pragma unroll
    for (int t = 0; t < 8; ++t)
      b[t] = ldfragB(bBase + (size_t)t * 128 * D1 + kc * 32);
#pragma unroll
    for (int t = 0; t < 8; ++t) {
      acc[0][t] = wmma32(a0, b[t], acc[0][t]);
      acc[1][t] = wmma32(a1, b[t], acc[1][t]);
    }
  }

  // ---- gates + r*cat epilogue ----
  const bool hasT = (i > 0), hasL = (j > 0), hasD = hasT && hasL;
  const int jm1 = hasL ? (j - 1) : 0;
  const float* pT = hbuf + (size_t)((i + 2) % 3) * ROWSZ + (size_t)j   * CELL;
  const float* pL = hbuf + (size_t)( i      % 3) * ROWSZ + (size_t)jm1 * CELL;
  const float* pD = hbuf + (size_t)((i + 2) % 3) * ROWSZ + (size_t)jm1 * CELL;
  _Float16* RCc = RC + (size_t)cell * BB * K2;
  float* Zc  = Z  + (size_t)cell * 4 * CELL;
  float* MIc = MI + (size_t)cell * CELL;
  const int uc = s * 16 + ucl;
#pragma unroll
  for (int mi = 0; mi < 2; ++mi) {
#pragma unroll
    for (int e = 0; e < 8; ++e) {
      int m = (mt0 + mi) * 16 + e + 8 * hs;
      float r0 = sigm(acc[mi][0][e] + br[uc]);
      float r1 = sigm(acc[mi][1][e] + br[128 + uc]);
      float r2 = sigm(acc[mi][2][e] + br[256 + uc]);
      float z0 = acc[mi][3][e] + bz[uc];          // zi
      float z1 = acc[mi][4][e] + bz[128 + uc];    // zl
      float z2 = acc[mi][5][e] + bz[256 + uc];    // zt
      float z3 = acc[mi][6][e] + bz[384 + uc];    // zd
      float zm = fmaxf(fmaxf(z0, z1), fmaxf(z2, z3));
      float e0 = __expf(z0 - zm), e1 = __expf(z1 - zm);
      float e2 = __expf(z2 - zm), e3 = __expf(z3 - zm);
      float inv = 1.0f / (e0 + e1 + e2 + e3);
      float hl = hasL ? pL[m * UU + uc] : 0.0f;
      float ht = hasT ? pT[m * UU + uc] : 0.0f;
      float hd = hasD ? pD[m * UU + uc] : 0.0f;
      // rc = r * [h_left | h_top | h_diag]
      RCc[m * K2 + uc]       = (_Float16)(r0 * hl);
      RCc[m * K2 + 128 + uc] = (_Float16)(r1 * ht);
      RCc[m * K2 + 256 + uc] = (_Float16)(r2 * hd);
      Zc[0 * CELL + m * UU + uc] = e0 * inv;
      Zc[1 * CELL + m * UU + uc] = e1 * inv;
      Zc[2 * CELL + m * UU + uc] = e2 * inv;
      Zc[3 * CELL + m * UU + uc] = e3 * inv;
      MIc[m * UU + uc] = acc[mi][7][e];           // s@w_ij
    }
  }
}

// ---------------- kernel C: GEMM2 + h update per cell ----------------
__global__ __launch_bounds__(256) void k_cellC(
    const _Float16* __restrict__ RC, const _Float16* __restrict__ UT,
    const float* __restrict__ MI, const float* __restrict__ Z,
    const float* __restrict__ bij, float* __restrict__ hbuf, int d, int i0) {
  const int cell = blockIdx.y;
  const int i = i0 + cell, j = d - i;
  const int tid = threadIdx.x, lane = tid & 31, wave = tid >> 5;
  const int mt = blockIdx.x * 8 + wave;
  const bool hasT = (i > 0), hasL = (j > 0), hasD = hasT && hasL;
  const int jm1 = hasL ? (j - 1) : 0;
  const float* pT = hbuf + (size_t)((i + 2) % 3) * ROWSZ + (size_t)j   * CELL;
  const float* pL = hbuf + (size_t)( i      % 3) * ROWSZ + (size_t)jm1 * CELL;
  const float* pD = hbuf + (size_t)((i + 2) % 3) * ROWSZ + (size_t)jm1 * CELL;
  float* pO       = hbuf + (size_t)( i      % 3) * ROWSZ + (size_t)j   * CELL;
  const _Float16* RCc = RC + (size_t)cell * BB * K2;
  const float* Zc  = Z  + (size_t)cell * 4 * CELL;
  const float* MIc = MI + (size_t)cell * CELL;
  const int hs = lane >> 4, ucl = lane & 15;

  const _Float16* aBase = RCc + (size_t)(mt * 16 + ucl) * K2 + hs * 16;
  const _Float16* bBase = UT + (size_t)ucl * K2 + hs * 16;

  f8 acc[8] = {};
#pragma unroll 1
  for (int kc = 0; kc < 12; ++kc) {
    h16 a = ldfragA(aBase + kc * 32);
    h16 b[8];
#pragma unroll
    for (int t = 0; t < 8; ++t)
      b[t] = ldfragB(bBase + (size_t)t * 16 * K2 + kc * 32);
#pragma unroll
    for (int t = 0; t < 8; ++t) acc[t] = wmma32(a, b[t], acc[t]);
  }
#pragma unroll
  for (int t = 0; t < 8; ++t) {
    int uc = t * 16 + ucl;
#pragma unroll
    for (int e = 0; e < 8; ++e) {
      int m = mt * 16 + e + 8 * hs;
      float hhat = tanhf(acc[t][e] + MIc[m * UU + uc] + bij[uc]);
      float zi = Zc[0 * CELL + m * UU + uc];
      float zl = Zc[1 * CELL + m * UU + uc];
      float zt = Zc[2 * CELL + m * UU + uc];
      float zd = Zc[3 * CELL + m * UU + uc];
      float hl = hasL ? pL[m * UU + uc] : 0.0f;
      float ht = hasT ? pT[m * UU + uc] : 0.0f;
      float hd = hasD ? pD[m * UU + uc] : 0.0f;
      pO[m * UU + uc] = zl * hl + zt * ht + zd * hd + zi * hhat;
    }
  }
}

__global__ void k_final(const float* __restrict__ hbuf, float* __restrict__ out) {
  int idx = blockIdx.x * 256 + threadIdx.x;
  // h at (L-1, R-1): row 63 -> buffer 63%3 == 0, col 63
  out[idx] = hbuf[(size_t)63 * CELL + idx];
}

// ---------------- host ----------------
extern "C" void kernel_launch(void* const* d_in, const int* in_sizes, int n_in,
                              void* d_out, int out_size, void* d_ws, size_t ws_size,
                              hipStream_t stream) {
  const float* x   = (const float*)d_in[0];
  const float* wr  = (const float*)d_in[1];
  const float* br  = (const float*)d_in[2];
  const float* wz  = (const float*)d_in[3];
  const float* bz  = (const float*)d_in[4];
  const float* wij = (const float*)d_in[5];
  const float* bij = (const float*)d_in[6];
  const float* U   = (const float*)d_in[7];

  char* ws = (char*)d_ws;
  _Float16* WT = (_Float16*)(ws + OFF_WT);
  _Float16* UT = (_Float16*)(ws + OFF_UT);
  float*    HB = (float*)(ws + OFF_HB);
  _Float16* RC = (_Float16*)(ws + OFF_RC);
  float*    Z  = (float*)(ws + OFF_Z);
  float*    MI = (float*)(ws + OFF_MI);
  _Float16* QF = (_Float16*)(ws + OFF_QF);

  k_prep_wt<<<(NN * D1 + 255) / 256, 256, 0, stream>>>(wr, wz, wij, WT);
  k_prep_ut<<<(UU * K2 + 255) / 256, 256, 0, stream>>>(U, UT);

  for (int d = 0; d < LL + RR - 1; ++d) {
    int i0 = (d > RR - 1) ? d - (RR - 1) : 0;
    int i1 = (d < LL - 1) ? d : LL - 1;
    int nc = i1 - i0 + 1;
    k_stageQ<<<dim3(4, nc), 256, 0, stream>>>(x, HB, QF, d, i0);
    k_cellA<<<dim3(8, nc), 256, 0, stream>>>(QF, WT, br, bz, HB, RC, Z, MI, d, i0);
    k_cellC<<<dim3(2, nc), 256, 0, stream>>>(RC, UT, MI, Z, bij, HB, d, i0);
  }
  k_final<<<(BB * UU + 255) / 256, 256, 0, stream>>>(HB, (float*)d_out);
}
